// EncoderText_44530220925149
// MI455X (gfx1250) — matrix operand order repaired and checked by reading, not compiled, not run on hardware
//
#include <hip/hip_runtime.h>
#include <hip/hip_bf16.h>

#define B_   128
#define T_   128
#define D_   300
#define DP   320
#define E_   1024
#define ASP  500
#define INVTEMP 1.25f
#define HALF_LOG2PI 0.91893853320467274f

// tiling
#define KC_C  10
#define NT_C  60
#define G3P   (NT_C*16)
#define KC_MU 19
#define NT_MU 64
#define MUPP  (NT_MU*16)
#define KC_RI 10
#define KC_RH 32
#define NT_R  192

typedef __attribute__((ext_vector_type(16))) __bf16         v16bf;
typedef __attribute__((ext_vector_type(16))) unsigned short v16u;
typedef __attribute__((ext_vector_type(8)))  float          v8f;

__device__ inline unsigned short f2bfbits(float f) {
  unsigned u = __builtin_bit_cast(unsigned, f);
  unsigned r = u + 0x7FFFu + ((u >> 16) & 1u);   // RNE to bf16
  return (unsigned short)(r >> 16);
}

// fragment = 1024 bytes: lane l owns 32 contiguous bytes (16 bf16)
__device__ inline v16bf load_frag(const unsigned short* __restrict__ P, int frag) {
  const v16u* p = (const v16u*)P + (size_t)frag * 32 + (threadIdx.x & 31);
  return __builtin_bit_cast(v16bf, *p);
}

// async copy of 16 bytes global->LDS, tracked by ASYNCcnt (CDNA5)
__device__ inline void async_cp16(unsigned lds_addr, const void* gaddr) {
  asm volatile("global_load_async_to_lds_b128 %0, %1, off"
               :: "v"(lds_addr), "v"(gaddr) : "memory");
}
__device__ inline void wait_async0() {
  asm volatile("s_wait_asynccnt 0x0" ::: "memory");
}

// read a B fragment from LDS: lane l reads its 32 contiguous bytes (2x ds_load_b128)
__device__ inline v16bf frag_lds(const unsigned short* sb) {
  const v16u* p = (const v16u*)sb + (threadIdx.x & 31);
  return __builtin_bit_cast(v16bf, *p);
}

// C store: lane l -> N=l&15 ; rows r=0..7 -> M = (l>>4)*8 + r
__device__ inline void store_c(float* __restrict__ C, int ldc, int m0, int n0, v8f c,
                               const float* __restrict__ bias, int nmax) {
  int lane  = threadIdx.x & 31;
  int n     = n0 + (lane & 15);
  int mbase = m0 + ((lane >> 4) << 3);
  float bv = (n < nmax) ? bias[n] : 0.0f;
#pragma unroll
  for (int r = 0; r < 8; ++r)
    C[(size_t)(mbase + r) * ldc + n] = c[r] + bv;
}

#define WMMA_BF16(ACC, A, Bf) \
  ACC = __builtin_amdgcn_wmma_f32_16x16x32_bf16(false, (A), false, (Bf), (short)0, (ACC), false, false)

// ---------- RNG ----------
__device__ inline unsigned pcg(unsigned x) {
  x = x * 747796405u + 2891336453u;
  unsigned w = ((x >> ((x >> 28) + 4u)) ^ x) * 277803737u;
  return (w >> 22) ^ w;
}
__device__ inline float u01(unsigned s) { return (pcg(s) >> 8) * (1.0f / 16777216.0f) + 5.9604645e-8f; }
__device__ inline float gumbelf(unsigned s) {
  float u = u01(s);
  return -__logf(-__logf(u + 1e-20f) + 1e-20f);
}
__device__ inline float normalf(unsigned s) {
  float u1 = u01(s * 2u + 1u), u2 = u01(s * 2u + 2u);
  return sqrtf(-2.0f * __logf(u1)) * __cosf(6.28318530718f * u2);
}

// ---------- pack kernels ----------
__global__ void pack_a(const float* __restrict__ src, int lda, int kmax, int KC, int do_relu,
                       unsigned short* __restrict__ dst, int total) {
  int idx = blockIdx.x * 256 + threadIdx.x;
  if (idx >= total) return;
  int e    = idx & 15;
  int lane = (idx >> 4) & 31;
  int fi   = idx >> 9;
  int kc   = fi % KC;
  int mt   = fi / KC;
  int row  = mt * 16 + (lane & 15);
  int half = lane >> 4;
  int vg   = e >> 1;
  int k    = kc * 32 + ((vg & 4) << 2) + half * 8 + ((vg & 3) << 1) + (e & 1);
  float f = (k < kmax) ? src[(size_t)row * lda + k] : 0.0f;
  if (do_relu) f = fmaxf(f, 0.0f);
  dst[idx] = f2bfbits(f);
}

__global__ void pack_cat(const float* __restrict__ xrow0, int ldx, const float* __restrict__ h,
                         unsigned short* __restrict__ dst, int total) {
  int idx = blockIdx.x * 256 + threadIdx.x;
  if (idx >= total) return;
  int e    = idx & 15;
  int lane = (idx >> 4) & 31;
  int fi   = idx >> 9;
  int kc   = fi % KC_MU;
  int mt   = fi / KC_MU;
  int row  = mt * 16 + (lane & 15);
  int half = lane >> 4;
  int vg   = e >> 1;
  int k    = kc * 32 + ((vg & 4) << 2) + half * 8 + ((vg & 3) << 1) + (e & 1);
  float f;
  if (k < D_)          f = fmaxf(xrow0[(size_t)row * ldx + k], 0.0f);
  else if (k < 2 * D_) f = fmaxf(h[(size_t)row * DP + (k - D_)], 0.0f);
  else                 f = 0.0f;
  dst[idx] = f2bfbits(f);
}

__global__ void pack_b(const float* __restrict__ src, unsigned short* __restrict__ dst,
                       int N, int K, int NT, int total) {
  int idx = blockIdx.x * 256 + threadIdx.x;
  if (idx >= total) return;
  int e    = idx & 15;
  int lane = (idx >> 4) & 31;
  int fi   = idx >> 9;
  int nt   = fi % NT;
  int kc   = fi / NT;
  int n    = nt * 16 + (lane & 15);
  int k    = kc * 32 + ((e & 8) << 1) + ((lane >> 4) << 3) + (e & 7);
  float v = (k < K && n < N) ? src[(size_t)n * K + k] : 0.0f;
  dst[idx] = f2bfbits(v);
}

__global__ void zero_f(float* p, int n) {
  int i = blockIdx.x * 256 + threadIdx.x;
  if (i < n) p[i] = 0.0f;
}

// ---------- GEMM kernels ----------
// cell: 8 waves (M tiles) x 4 N-tiles; B(ih,hh) double-buffered in LDS via async copy
__global__ void cell_gemm(const unsigned short* __restrict__ Ax, const unsigned short* __restrict__ Ah,
                          const unsigned short* __restrict__ Bih, const unsigned short* __restrict__ Bhh,
                          const float* __restrict__ bih, const float* __restrict__ bhh,
                          float* __restrict__ gi, float* __restrict__ gh) {
  __shared__ unsigned short sB[2][8][512];   // [stage][frag 0-3:Bih, 4-7:Bhh][512 bf16]
  int tid = threadIdx.x;
  int wv  = tid >> 5;
  int m0  = wv * 16;
  int ntb = blockIdx.x * 4;

  auto issue = [&](int st, int kc) {
    // 8 KB per stage: 2 x (256 threads * 16B)
#pragma unroll
    for (int r = 0; r < 2; ++r) {
      int off = (r * 256 + tid) * 16;         // byte offset within stage
      int f   = off >> 10;                     // fragment 0..7
      int fo  = off & 1023;                    // byte within fragment
      const unsigned short* g = (f < 4)
          ? Bih + (size_t)(kc * NT_C + ntb + f) * 512 + (fo >> 1)
          : Bhh + (size_t)(kc * NT_C + ntb + (f - 4)) * 512 + (fo >> 1);
      unsigned laddr = (unsigned)(size_t)(&sB[st][0][0]) + (unsigned)off;
      async_cp16(laddr, g);
    }
  };

  v8f aI0 = {}, aI1 = {}, aI2 = {}, aI3 = {};
  v8f aH0 = {}, aH1 = {}, aH2 = {}, aH3 = {};

  issue(0, 0);
  wait_async0();
  __syncthreads();
  for (int kc = 0; kc < KC_C; ++kc) {
    int st = kc & 1;
    if (kc + 1 < KC_C) issue(st ^ 1, kc + 1);
    v16bf ax = load_frag(Ax, wv * KC_C + kc);
    v16bf ah = load_frag(Ah, wv * KC_C + kc);
    v16bf b0 = frag_lds(sB[st][0]); WMMA_BF16(aI0, ax, b0);
    v16bf b1 = frag_lds(sB[st][1]); WMMA_BF16(aI1, ax, b1);
    v16bf b2 = frag_lds(sB[st][2]); WMMA_BF16(aI2, ax, b2);
    v16bf b3 = frag_lds(sB[st][3]); WMMA_BF16(aI3, ax, b3);
    v16bf c0 = frag_lds(sB[st][4]); WMMA_BF16(aH0, ah, c0);
    v16bf c1 = frag_lds(sB[st][5]); WMMA_BF16(aH1, ah, c1);
    v16bf c2 = frag_lds(sB[st][6]); WMMA_BF16(aH2, ah, c2);
    v16bf c3 = frag_lds(sB[st][7]); WMMA_BF16(aH3, ah, c3);
    if (kc + 1 < KC_C) { wait_async0(); __syncthreads(); }
  }
  store_c(gi, G3P, m0, (ntb + 0) * 16, aI0, bih, 3 * D_);
  store_c(gi, G3P, m0, (ntb + 1) * 16, aI1, bih, 3 * D_);
  store_c(gi, G3P, m0, (ntb + 2) * 16, aI2, bih, 3 * D_);
  store_c(gi, G3P, m0, (ntb + 3) * 16, aI3, bih, 3 * D_);
  store_c(gh, G3P, m0, (ntb + 0) * 16, aH0, bhh, 3 * D_);
  store_c(gh, G3P, m0, (ntb + 1) * 16, aH1, bhh, 3 * D_);
  store_c(gh, G3P, m0, (ntb + 2) * 16, aH2, bhh, 3 * D_);
  store_c(gh, G3P, m0, (ntb + 3) * 16, aH3, bhh, 3 * D_);
}

__global__ void mu_gemm(const unsigned short* __restrict__ Acat, const unsigned short* __restrict__ Bmu,
                        const float* __restrict__ bmu, float* __restrict__ mu) {
  __shared__ unsigned short sB[2][4][512];
  int tid = threadIdx.x;
  int wv  = tid >> 5;
  int m0  = wv * 16;
  int ntb = blockIdx.x * 4;

  auto issue = [&](int st, int kc) {
    int off = tid * 16;                        // 4 KB per stage: one 16B per thread
    int f   = off >> 10;
    int fo  = off & 1023;
    const unsigned short* g = Bmu + (size_t)(kc * NT_MU + ntb + f) * 512 + (fo >> 1);
    unsigned laddr = (unsigned)(size_t)(&sB[st][0][0]) + (unsigned)off;
    async_cp16(laddr, g);
  };

  v8f a0 = {}, a1 = {}, a2 = {}, a3 = {};
  issue(0, 0);
  wait_async0();
  __syncthreads();
  for (int kc = 0; kc < KC_MU; ++kc) {
    int st = kc & 1;
    if (kc + 1 < KC_MU) issue(st ^ 1, kc + 1);
    v16bf a = load_frag(Acat, wv * KC_MU + kc);
    v16bf b0 = frag_lds(sB[st][0]); WMMA_BF16(a0, a, b0);
    v16bf b1 = frag_lds(sB[st][1]); WMMA_BF16(a1, a, b1);
    v16bf b2 = frag_lds(sB[st][2]); WMMA_BF16(a2, a, b2);
    v16bf b3 = frag_lds(sB[st][3]); WMMA_BF16(a3, a, b3);
    if (kc + 1 < KC_MU) { wait_async0(); __syncthreads(); }
  }
  store_c(mu, MUPP, m0, (ntb + 0) * 16, a0, bmu, 2 * ASP);
  store_c(mu, MUPP, m0, (ntb + 1) * 16, a1, bmu, 2 * ASP);
  store_c(mu, MUPP, m0, (ntb + 2) * 16, a2, bmu, 2 * ASP);
  store_c(mu, MUPP, m0, (ntb + 3) * 16, a3, bmu, 2 * ASP);
}

__global__ void rnn_gemm(const unsigned short* __restrict__ Atext, const unsigned short* __restrict__ Ahr,
                         const unsigned short* __restrict__ Bih, const unsigned short* __restrict__ Bhh,
                         const float* __restrict__ bih, const float* __restrict__ bhh,
                         float* __restrict__ gi, float* __restrict__ gh) {
  __shared__ unsigned short sB[2][4][512];
  int tid = threadIdx.x;
  int wv  = tid >> 5;
  int m0  = wv * 16;
  int ntb = blockIdx.x * 4;

  auto issue = [&](const unsigned short* Bsrc, int st, int kc) {
    int off = tid * 16;
    int f   = off >> 10;
    int fo  = off & 1023;
    const unsigned short* g = Bsrc + (size_t)(kc * NT_R + ntb + f) * 512 + (fo >> 1);
    unsigned laddr = (unsigned)(size_t)(&sB[st][0][0]) + (unsigned)off;
    async_cp16(laddr, g);
  };

  // pass 1: gi = text @ Wih^T + bih
  {
    v8f a0 = {}, a1 = {}, a2 = {}, a3 = {};
    issue(Bih, 0, 0);
    wait_async0();
    __syncthreads();
    for (int kc = 0; kc < KC_RI; ++kc) {
      int st = kc & 1;
      if (kc + 1 < KC_RI) issue(Bih, st ^ 1, kc + 1);
      v16bf a = load_frag(Atext, wv * KC_RI + kc);
      v16bf b0 = frag_lds(sB[st][0]); WMMA_BF16(a0, a, b0);
      v16bf b1 = frag_lds(sB[st][1]); WMMA_BF16(a1, a, b1);
      v16bf b2 = frag_lds(sB[st][2]); WMMA_BF16(a2, a, b2);
      v16bf b3 = frag_lds(sB[st][3]); WMMA_BF16(a3, a, b3);
      if (kc + 1 < KC_RI) { wait_async0(); __syncthreads(); }
    }
    store_c(gi, 3 * E_, m0, (ntb + 0) * 16, a0, bih, 3 * E_);
    store_c(gi, 3 * E_, m0, (ntb + 1) * 16, a1, bih, 3 * E_);
    store_c(gi, 3 * E_, m0, (ntb + 2) * 16, a2, bih, 3 * E_);
    store_c(gi, 3 * E_, m0, (ntb + 3) * 16, a3, bih, 3 * E_);
  }
  __syncthreads();   // all waves done reading pass-1 buffers before pass-2 overwrites
  // pass 2: gh = h_rnn @ Whh^T + bhh
  {
    v8f c0 = {}, c1 = {}, c2 = {}, c3 = {};
    issue(Bhh, 0, 0);
    wait_async0();
    __syncthreads();
    for (int kc = 0; kc < KC_RH; ++kc) {
      int st = kc & 1;
      if (kc + 1 < KC_RH) issue(Bhh, st ^ 1, kc + 1);
      v16bf a = load_frag(Ahr, wv * KC_RH + kc);
      v16bf b0 = frag_lds(sB[st][0]); WMMA_BF16(c0, a, b0);
      v16bf b1 = frag_lds(sB[st][1]); WMMA_BF16(c1, a, b1);
      v16bf b2 = frag_lds(sB[st][2]); WMMA_BF16(c2, a, b2);
      v16bf b3 = frag_lds(sB[st][3]); WMMA_BF16(c3, a, b3);
      if (kc + 1 < KC_RH) { wait_async0(); __syncthreads(); }
    }
    store_c(gh, 3 * E_, m0, (ntb + 0) * 16, c0, bhh, 3 * E_);
    store_c(gh, 3 * E_, m0, (ntb + 1) * 16, c1, bhh, 3 * E_);
    store_c(gh, 3 * E_, m0, (ntb + 2) * 16, c2, bhh, 3 * E_);
    store_c(gh, 3 * E_, m0, (ntb + 3) * 16, c3, bhh, 3 * E_);
  }
}

// ---------- elementwise kernels ----------
__global__ void cell_update(const float* __restrict__ gi, const float* __restrict__ gh,
                            float* __restrict__ h) {
  int idx = blockIdx.x * 256 + threadIdx.x;
  if (idx >= B_ * D_) return;
  int b = idx / D_, d = idx % D_;
  const float* gir = gi + (size_t)b * G3P;
  const float* ghr = gh + (size_t)b * G3P;
  float r = 1.0f / (1.0f + __expf(-(gir[d] + ghr[d])));
  float z = 1.0f / (1.0f + __expf(-(gir[d + D_] + ghr[d + D_])));
  float n = tanhf(gir[d + 2 * D_] + r * ghr[d + 2 * D_]);
  float* hp = h + (size_t)b * DP + d;
  *hp = (1.0f - z) * n + z * (*hp);
}

__global__ void sample_kernel(const float* __restrict__ x, const float* __restrict__ h,
                              const float* __restrict__ mu,
                              const float* __restrict__ Wlv, const float* __restrict__ blv,
                              float* __restrict__ text, float* __restrict__ out, int t) {
  int bh = blockIdx.x, b = bh >> 1, head = bh & 1;
  int tid = threadIdx.x;
  __shared__ float red[128];
  __shared__ int   redi[128];
  __shared__ float attsh;

  const float* xr = x + ((size_t)b * T_ + t) * D_;
  const float* hr = h + (size_t)b * DP;

  float acc = 0.0f;
  for (int k = tid; k < 2 * D_; k += 128) {
    float c = (k < D_) ? fmaxf(xr[k], 0.0f) : fmaxf(hr[k - D_], 0.0f);
    acc += c * Wlv[(size_t)head * 2 * D_ + k];
  }
  red[tid] = acc; __syncthreads();
  for (int s = 64; s > 0; s >>= 1) { if (tid < s) red[tid] += red[tid + s]; __syncthreads(); }
  float lvr = red[0] + blv[head];
  float lv = (lvr > 20.0f) ? lvr : log1pf(__expf(lvr));
  __syncthreads();

  const float* murow = mu + (size_t)b * MUPP + head * ASP;
  unsigned gbase = ((unsigned)t * 131u + (unsigned)bh) * 2654435761u;
  unsigned cbase = gbase + 1000003u;
  float lvals[4], lmax = -1e30f;
#pragma unroll
  for (int i = 0; i < 4; ++i) {
    int a = tid + i * 128;
    float l = (a < ASP) ? (murow[a] + gumbelf(gbase + (unsigned)a)) * INVTEMP : -1e30f;
    lvals[i] = l; lmax = fmaxf(lmax, l);
  }
  red[tid] = lmax; __syncthreads();
  for (int s = 64; s > 0; s >>= 1) { if (tid < s) red[tid] = fmaxf(red[tid], red[tid + s]); __syncthreads(); }
  lmax = red[0]; __syncthreads();

  float se = 0.0f;
#pragma unroll
  for (int i = 0; i < 4; ++i) { int a = tid + i * 128; if (a < ASP) se += __expf(lvals[i] - lmax); }
  red[tid] = se; __syncthreads();
  for (int s = 64; s > 0; s >>= 1) { if (tid < s) red[tid] += red[tid + s]; __syncthreads(); }
  float S = red[0]; __syncthreads();

  float bestv = -1e30f; int besti = 0;
#pragma unroll
  for (int i = 0; i < 4; ++i) {
    int a = tid + i * 128;
    if (a < ASP) {
      float v = lvals[i] + gumbelf(cbase + (unsigned)a);
      if (v > bestv) { bestv = v; besti = a; }
    }
  }
  red[tid] = bestv; redi[tid] = besti; __syncthreads();
  for (int s = 64; s > 0; s >>= 1) {
    if (tid < s && red[tid + s] > red[tid]) { red[tid] = red[tid + s]; redi[tid] = redi[tid + s]; }
    __syncthreads();
  }
  int ind = redi[0];

  if (tid == 0) {
    float l_ind = (murow[ind] + gumbelf(gbase + (unsigned)ind)) * INVTEMP;
    float prob  = __expf(l_ind - lmax) / S;
    float action = (float)ind / (float)ASP;
    float stdv = __expf(0.5f * lv);
    float e1 = normalf(gbase + 2000003u);
    float e2 = normalf(gbase + 2000007u);
    float s1 = action + stdv * e1;
    float logp = -0.5f * e2 * e2 - 0.5f * lv - HALF_LOG2PI;
    float att = 20.0f / (1.0f + __expf(-s1));
    out[131072 + (size_t)b * 256 + t * 2 + head]   = __logf(prob);
    out[163840 + (size_t)b * 256 + head * 128 + t] = action;
    out[196608 + (size_t)b * 256 + head * 128 + t] = logp;
    attsh = att;
  }
  __syncthreads();
  float att = attsh;
  for (int j = tid; j < D_ / 2; j += 128)
    text[(size_t)b * DP + head * (D_ / 2) + j] = xr[head * (D_ / 2) + j] * att;
}

__global__ void rnn_update(const float* __restrict__ gi, const float* __restrict__ gh,
                           float* __restrict__ h, float* __restrict__ last,
                           const int* __restrict__ lengths, int t) {
  int idx = blockIdx.x * 256 + threadIdx.x;
  if (idx >= B_ * E_) return;
  int b = idx >> 10, d = idx & (E_ - 1);
  const float* gir = gi + (size_t)b * 3 * E_;
  const float* ghr = gh + (size_t)b * 3 * E_;
  float r = 1.0f / (1.0f + __expf(-(gir[d] + ghr[d])));
  float z = 1.0f / (1.0f + __expf(-(gir[d + E_] + ghr[d + E_])));
  float n = tanhf(gir[d + 2 * E_] + r * ghr[d + 2 * E_]);
  float hv = h[idx];
  float hn = (1.0f - z) * n + z * hv;
  h[idx] = hn;
  if (t == lengths[b] - 1) last[idx] = hn;
}

__global__ void bn_kernel(const float* __restrict__ last, const float* __restrict__ gamma,
                          const float* __restrict__ beta, float* __restrict__ out) {
  int f = blockIdx.x, b = threadIdx.x;
  __shared__ float red[128];
  float v = last[(size_t)b * E_ + f];
  red[b] = v; __syncthreads();
  for (int s = 64; s > 0; s >>= 1) { if (b < s) red[b] += red[b + s]; __syncthreads(); }
  float mean = red[0] * (1.0f / B_); __syncthreads();
  float dv = v - mean;
  red[b] = dv * dv; __syncthreads();
  for (int s = 64; s > 0; s >>= 1) { if (b < s) red[b] += red[b + s]; __syncthreads(); }
  float var = red[0] * (1.0f / B_);
  out[(size_t)b * E_ + f] = gamma[f] * dv * rsqrtf(var + 1e-5f) + beta[f];
}

// ---------- launch ----------
extern "C" void kernel_launch(void* const* d_in, const int* in_sizes, int n_in,
                              void* d_out, int out_size, void* d_ws, size_t ws_size,
                              hipStream_t stream) {
  const float* x        = (const float*)d_in[0];
  const int*   lengths  = (const int*)d_in[1];
  const float* Wih_cell = (const float*)d_in[3];
  const float* Whh_cell = (const float*)d_in[4];
  const float* bih_cell = (const float*)d_in[5];
  const float* bhh_cell = (const float*)d_in[6];
  const float* Wmu      = (const float*)d_in[7];
  const float* bmu      = (const float*)d_in[8];
  const float* Wlv      = (const float*)d_in[9];
  const float* blv      = (const float*)d_in[10];
  const float* Wih_rnn  = (const float*)d_in[11];
  const float* Whh_rnn  = (const float*)d_in[12];
  const float* bih_rnn  = (const float*)d_in[13];
  const float* bhh_rnn  = (const float*)d_in[14];
  const float* bn_g     = (const float*)d_in[15];
  const float* bn_b     = (const float*)d_in[16];
  float* out = (float*)d_out;

  char* ws = (char*)d_ws;
  size_t off = 0;
  auto alloc = [&](size_t bytes) { size_t o = off; off = (off + bytes + 255) & ~(size_t)255; return o; };

  unsigned short* wBihC = (unsigned short*)(ws + alloc((size_t)KC_C  * NT_C  * 1024));
  unsigned short* wBhhC = (unsigned short*)(ws + alloc((size_t)KC_C  * NT_C  * 1024));
  unsigned short* wBmu  = (unsigned short*)(ws + alloc((size_t)KC_MU * NT_MU * 1024));
  unsigned short* wBihR = (unsigned short*)(ws + alloc((size_t)KC_RI * NT_R  * 1024));
  unsigned short* wBhhR = (unsigned short*)(ws + alloc((size_t)KC_RH * NT_R  * 1024));
  unsigned short* Ax    = (unsigned short*)(ws + alloc((size_t)8 * KC_C  * 1024));
  unsigned short* AhC   = (unsigned short*)(ws + alloc((size_t)8 * KC_C  * 1024));
  unsigned short* Acat  = (unsigned short*)(ws + alloc((size_t)8 * KC_MU * 1024));
  unsigned short* Atx   = (unsigned short*)(ws + alloc((size_t)8 * KC_RI * 1024));
  unsigned short* AhR   = (unsigned short*)(ws + alloc((size_t)8 * KC_RH * 1024));
  float* h_cell = (float*)(ws + alloc((size_t)B_ * DP * 4));
  float* h_rnn  = (float*)(ws + alloc((size_t)B_ * E_ * 4));
  float* text   = (float*)(ws + alloc((size_t)B_ * DP * 4));
  float* lastb  = (float*)(ws + alloc((size_t)B_ * E_ * 4));
  float* gi_c   = (float*)(ws + alloc((size_t)B_ * G3P * 4));
  float* gh_c   = (float*)(ws + alloc((size_t)B_ * G3P * 4));
  float* mu     = (float*)(ws + alloc((size_t)B_ * MUPP * 4));
  float* gi_r   = (float*)(ws + alloc((size_t)B_ * 3 * E_ * 4));
  float* gh_r   = (float*)(ws + alloc((size_t)B_ * 3 * E_ * 4));

  auto cdiv = [](int a, int b) { return (a + b - 1) / b; };

  { int tot = KC_C * NT_C * 512;
    pack_b<<<cdiv(tot, 256), 256, 0, stream>>>(Wih_cell, wBihC, 3 * D_, D_, NT_C, tot);
    pack_b<<<cdiv(tot, 256), 256, 0, stream>>>(Whh_cell, wBhhC, 3 * D_, D_, NT_C, tot); }
  { int tot = KC_MU * NT_MU * 512;
    pack_b<<<cdiv(tot, 256), 256, 0, stream>>>(Wmu, wBmu, 2 * ASP, 2 * D_, NT_MU, tot); }
  { int tot = KC_RI * NT_R * 512;
    pack_b<<<cdiv(tot, 256), 256, 0, stream>>>(Wih_rnn, wBihR, 3 * E_, D_, NT_R, tot); }
  { int tot = KC_RH * NT_R * 512;
    pack_b<<<cdiv(tot, 256), 256, 0, stream>>>(Whh_rnn, wBhhR, 3 * E_, E_, NT_R, tot); }

  zero_f<<<cdiv(B_ * DP, 256), 256, 0, stream>>>(h_cell, B_ * DP);
  zero_f<<<cdiv(B_ * E_, 256), 256, 0, stream>>>(h_rnn, B_ * E_);
  zero_f<<<cdiv(B_ * E_, 256), 256, 0, stream>>>(lastb, B_ * E_);
  zero_f<<<cdiv(8 * KC_C * 256, 256), 256, 0, stream>>>((float*)AhC, 8 * KC_C * 256);
  zero_f<<<cdiv(8 * KC_RH * 256, 256), 256, 0, stream>>>((float*)AhR, 8 * KC_RH * 256);

  const int totAx   = 8 * KC_C  * 512;
  const int totAcat = 8 * KC_MU * 512;
  const int totAhR  = 8 * KC_RH * 512;

  for (int t = 0; t < T_; ++t) {
    pack_a<<<cdiv(totAx, 256), 256, 0, stream>>>(x + (size_t)t * D_, T_ * D_, D_, KC_C, 0, Ax, totAx);
    cell_gemm<<<NT_C / 4, 256, 0, stream>>>(Ax, AhC, wBihC, wBhhC, bih_cell, bhh_cell, gi_c, gh_c);
    cell_update<<<cdiv(B_ * D_, 256), 256, 0, stream>>>(gi_c, gh_c, h_cell);
    pack_a<<<cdiv(totAx, 256), 256, 0, stream>>>(h_cell, DP, D_, KC_C, 0, AhC, totAx);
    pack_cat<<<cdiv(totAcat, 256), 256, 0, stream>>>(x + (size_t)t * D_, T_ * D_, h_cell, Acat, totAcat);
    mu_gemm<<<NT_MU / 4, 256, 0, stream>>>(Acat, wBmu, bmu, mu);
    sample_kernel<<<B_ * 2, 128, 0, stream>>>(x, h_cell, mu, Wlv, blv, text, out, t);
    pack_a<<<cdiv(totAx, 256), 256, 0, stream>>>(text, DP, D_, KC_RI, 0, Atx, totAx);
    rnn_gemm<<<NT_R / 4, 256, 0, stream>>>(Atx, AhR, wBihR, wBhhR, bih_rnn, bhh_rnn, gi_r, gh_r);
    rnn_update<<<cdiv(B_ * E_, 256), 256, 0, stream>>>(gi_r, gh_r, h_rnn, lastb, lengths, t);
    pack_a<<<cdiv(totAhR, 256), 256, 0, stream>>>(h_rnn, E_, E_, KC_RH, 0, AhR, totAhR);
  }

  bn_kernel<<<E_, 128, 0, stream>>>(lastb, bn_g, bn_b, out);
}
